// snn_model_41583873360170
// MI455X (gfx1250) — matrix operand order repaired and checked by reading, not compiled, not run on hardware
//
#include <hip/hip_runtime.h>

// ---------------------------------------------------------------------------
// SNN over 64 timesteps, refactored into:
//   (1) deterministic input-spike precompute (elementwise),
//   (2) per-layer skinny GEMM (64 x K x N) with bf16 WMMA, hi/lo-split f32
//       weights for ~f32 accuracy, fused with the 64-step membrane scan.
// Weight tiles (the 336 MB roofline stream) are DMA'd into LDS by the CDNA5
// Tensor Data Mover (tensor_load_to_lds + s_wait_tensorcnt), double-buffered.
// Each layer reads its A-matrix (spike rows, 0/1 as f32) straight from the
// previous layer's region of d_out: the pad_head layout is exactly the
// pipeline-delayed spike input the next layer needs.
// ---------------------------------------------------------------------------

typedef unsigned short u16;
typedef unsigned int   u32;
typedef __attribute__((ext_vector_type(16))) __bf16 v16bf;
typedef __attribute__((ext_vector_type(16))) u16    v16u;
typedef __attribute__((ext_vector_type(8)))  float  v8f;
typedef __attribute__((ext_vector_type(4)))  float  f32x4;
typedef __attribute__((ext_vector_type(4)))  u32    u32x4;
typedef __attribute__((ext_vector_type(8)))  int    i32x8;
typedef __attribute__((ext_vector_type(4)))  int    i32x4;

#if defined(__has_builtin)
#if __has_builtin(__builtin_amdgcn_tensor_load_to_lds) && \
    __has_builtin(__builtin_amdgcn_s_wait_tensorcnt)
#define SNN_HAVE_TDM 1
#endif
#endif
#ifndef SNN_HAVE_TDM
#define SNN_HAVE_TDM 0
#endif

#define BROWS 32                 // K rows per staged weight tile
#define BCOLS 64                 // columns per workgroup
#define BPAD  68                 // LDS row stride in floats (TDM pad: 64 + 4)

static __device__ __forceinline__ u16 bf16_trunc(float x) {
  // round-toward-zero f32 -> bf16 (exact for 0/1 spikes; fine for the lo
  // part of the hi/lo weight split)
  return (u16)(__builtin_bit_cast(unsigned int, x) >> 16);
}
static __device__ __forceinline__ v16bf as_bf16(v16u v) {
  return __builtin_bit_cast(v16bf, v);
}
static __device__ __forceinline__ unsigned lds_offset_of(const void* p) {
  // low 32 bits of a generic pointer into LDS == LDS byte offset
  return (unsigned)(unsigned long long)(uintptr_t)p;
}

// ---------------------------------------------------------------------------
// Kernel 1: input layer. m0 += img; fire at >=1; subtract 1 (no decay).
// Writes s0 rows 0..63 plus the zero pad row 64 (pad_tail layout).
// ---------------------------------------------------------------------------
__global__ __launch_bounds__(256) void snn_input_spikes(
    const float* __restrict__ img, float* __restrict__ out0, int n)
{
  int i = blockIdx.x * blockDim.x + threadIdx.x;
  if (i >= n) return;
  float x = img[i];
  float m = 0.0f;
  for (int t = 0; t < 64; ++t) {
    m += x;
    bool s = (m >= 1.0f);
    m = s ? (m - 1.0f) : m;
    out0[(size_t)t * n + i] = s ? 1.0f : 0.0f;
  }
  out0[(size_t)64 * n + i] = 0.0f;  // pad_tail zero row
}

#if SNN_HAVE_TDM
// ---------------------------------------------------------------------------
// TDM: DMA a BROWS x BCOLS f32 tile of row-major W (K x N) into LDS.
// Descriptor packed per the CDNA5 D# tables. Pad feature inserts 4 dwords
// after every 64 dwords (one tile row) -> LDS row stride = BPAD floats.
// Must be issued from a single wave (TDM ignores EXEC; guard on wave id).
// ---------------------------------------------------------------------------
static __device__ __forceinline__ void tdm_load_b_tile(
    const float* gsrc,      // &W[k0*N + colbase]  (tile origin, in bounds)
    unsigned     lds_byte,  // LDS byte offset of destination buffer
    int N, int K)
{
  unsigned long long ga = (unsigned long long)(uintptr_t)gsrc;
  u32x4 g0;
  g0[0] = 1u;                                    // count=1, no gather
  g0[1] = lds_byte;                              // lds_addr  [63:32]
  g0[2] = (u32)(ga & 0xFFFFFFFFull);             // global_addr[31:0]
  g0[3] = (u32)((ga >> 32) & 0x01FFFFFFull)      // global_addr[56:32]
        | (2u << 30);                            // type=2 ("image")
  i32x8 g1;
  g1[0] = (int)((2u << 16)                       // data_size = 4 bytes
              | (1u << 20)                       // pad_enable
              | (5u << 22)                       // pad_interval: 64 dwords
              | (3u << 25));                     // pad_amount:   4 dwords
  g1[1] = (int)(((u32)N & 0xFFFFu) << 16);                       // tdim0 lo
  g1[2] = (int)(((u32)N >> 16) | (((u32)K & 0xFFFFu) << 16));    // tdim0 hi | tdim1 lo
  g1[3] = (int)(((u32)K >> 16) | ((u32)BCOLS << 16));            // tdim1 hi | tile_dim0
  g1[4] = BROWS;                                 // tile_dim1 (tile_dim2 = 0)
  g1[5] = N;                                     // tensor_dim0_stride[31:0]
  g1[6] = 0;                                     // stride hi | dim1_stride lo
  g1[7] = 0;
  i32x4 z4 = {0, 0, 0, 0};
  i32x8 z8 = {0, 0, 0, 0, 0, 0, 0, 0};
  // 6-arg toolchain form: (g0, g1, g2, g3, extra, cpol)
  __builtin_amdgcn_tensor_load_to_lds(g0, g1, z4, z4, z8, 0);
}
#endif

// ---------------------------------------------------------------------------
// Kernel 2: one linear SNN layer.
//   A   : 64 x K spike matrix (f32 0/1 rows; already pipeline-shifted)
//   W   : K x N f32 weights (row-major)
//   Out : 65 x N pad_head spike output (row 0 zero, rows 1..64 = fires)
// Block = 128 threads = 4 waves; each wave owns a 64x16 C slab:
// 4 M-tiles x v_wmma_f32_16x16x32_bf16, x2 for the hi/lo weight split.
// ---------------------------------------------------------------------------
__global__ __launch_bounds__(128) void snn_layer_wmma(
    const float* __restrict__ A,
    const float* __restrict__ W,
    float* __restrict__ Out,
    int K, int N)
{
  __shared__ float lds_b[2][BROWS * BPAD];   // double-buffered weight tiles
  __shared__ float lds_c[4][64 * 16];        // per-wave C slab for the scan

  const int lane = threadIdx.x & 31;
  const int wave = threadIdx.x >> 5;
  const int ln   = lane & 15;                // column / row-within-tile index
  const int hi   = lane >> 4;                // half-wave select
  const int tilecol0 = blockIdx.x * BCOLS;   // first column of this WG
  const int colbase  = tilecol0 + wave * 16; // first column of this wave

  v8f c[4] = {};                             // 4 M-tiles of 16x16 f32 acc

  const int abase = hi * 8;                  // A: lanes 16-31 hold K+8/K+24
  const int kb    = hi * 8;                  // B: lanes 16-31 hold K+8/K+24

#if SNN_HAVE_TDM
  if (wave == 0) {
    tdm_load_b_tile(W + tilecol0, lds_offset_of(&lds_b[0][0]), N, K);
  }
#endif

  for (int k0 = 0; k0 < K; k0 += BROWS) {
    const int cur = (k0 >> 5) & 1;

#if SNN_HAVE_TDM
    if (wave == 0) {
      if (k0 + BROWS < K) {
        tdm_load_b_tile(W + (size_t)(k0 + BROWS) * N + tilecol0,
                        lds_offset_of(&lds_b[cur ^ 1][0]), N, K);
        __builtin_amdgcn_s_wait_tensorcnt(1);  // current tile landed
      } else {
        __builtin_amdgcn_s_wait_tensorcnt(0);
      }
    }
    __syncthreads();                           // publish tile to all waves
#else
    // Fallback: cooperative staging of the 32x64 weight tile.
    for (int i = threadIdx.x; i < BROWS * BCOLS; i += 128) {
      int r = i >> 6, cc = i & 63;
      lds_b[cur][r * BPAD + cc] = W[(size_t)(k0 + r) * N + tilecol0 + cc];
    }
    __builtin_prefetch(W + (size_t)(k0 + BROWS) * N + tilecol0 + threadIdx.x, 0, 0);
    __syncthreads();
#endif

    // ---- A fragments: issue all 16 global b128 loads into independent
    // temps first so they overlap, then convert (spikes 0/1: trunc exact).
    f32x4 pa[4][4];
#pragma unroll
    for (int mt = 0; mt < 4; ++mt) {
      const float* ar = A + (size_t)(mt * 16 + ln) * K + k0 + abase;
      pa[mt][0] = *(const f32x4*)(ar + 0);     // K = base+0..3
      pa[mt][1] = *(const f32x4*)(ar + 4);     // K = base+4..7
      pa[mt][2] = *(const f32x4*)(ar + 16);    // K = base+16..19
      pa[mt][3] = *(const f32x4*)(ar + 20);    // K = base+20..23
    }

    // ---- B fragment from LDS: column colbase+ln, K-runs of 8 at k0+kb and k0+kb+16.
    // hi/lo split: w == bf16_hi(w) + bf16(w - bf16_hi(w))  (exact in f32)
    v16u bhu, blu;
#pragma unroll
    for (int h = 0; h < 16; ++h) {
      float w   = lds_b[cur][(kb + (h & 7) + ((h >> 3) << 4)) * BPAD + wave * 16 + ln];
      unsigned wu = __builtin_bit_cast(unsigned int, w);
      float whi = __builtin_bit_cast(float, wu & 0xFFFF0000u);
      bhu[h] = (u16)(wu >> 16);
      blu[h] = bf16_trunc(w - whi);
    }
    v16bf bh = as_bf16(bhu);
    v16bf bl = as_bf16(blu);

#pragma unroll
    for (int mt = 0; mt < 4; ++mt) {
      v16u au;
#pragma unroll
      for (int j = 0; j < 4; ++j) {
        au[j]      = bf16_trunc(pa[mt][0][j]);
        au[4 + j]  = bf16_trunc(pa[mt][1][j]);
        au[8 + j]  = bf16_trunc(pa[mt][2][j]);
        au[12 + j] = bf16_trunc(pa[mt][3][j]);
      }
      v16bf a = as_bf16(au);
      c[mt] = __builtin_amdgcn_wmma_f32_16x16x32_bf16(
          false, a, false, bh, (short)0, c[mt], false, false);
      c[mt] = __builtin_amdgcn_wmma_f32_16x16x32_bf16(
          false, a, false, bl, (short)0, c[mt], false, false);
    }

    __syncthreads();   // all reads of the other buffer done before re-DMA
  }

  // ---- Dump C slab (rows = timesteps 0..63, cols = 16 neurons) to LDS.
  float* l = lds_c[wave];
#pragma unroll
  for (int mt = 0; mt < 4; ++mt)
#pragma unroll
    for (int r = 0; r < 8; ++r)
      l[(mt * 16 + r + hi * 8) * 16 + ln] = c[mt][r];
  __syncthreads();

  // ---- Membrane scan: lanes 0..15 each own one output column.
  if (lane < 16) {
    const int col = colbase + lane;
    Out[col] = 0.0f;  // pad_head row 0
    float m = 0.0f;
    for (int t = 0; t < 64; ++t) {
      m += l[t * 16 + lane];
      bool f = (m >= 1.0f);
      m = f ? (m - 1.0f) : (m * 0.5f);   // DECAY_MULT = 0.5
      Out[(size_t)(t + 1) * N + col] = f ? 1.0f : 0.0f;
    }
  }
}

// ---------------------------------------------------------------------------
// Launch: spikes -> layer1 -> layer2 -> layer3, all on `stream`.
// d_out layout (floats): [65x16384 | 65x4096 | 65x4096 | 65x1024]
// ---------------------------------------------------------------------------
extern "C" void kernel_launch(void* const* d_in, const int* in_sizes, int n_in,
                              void* d_out, int out_size, void* d_ws, size_t ws_size,
                              hipStream_t stream) {
  (void)in_sizes; (void)n_in; (void)out_size; (void)d_ws; (void)ws_size;

  const float* img = (const float*)d_in[0];
  const float* w1  = (const float*)d_in[1];
  const float* w2  = (const float*)d_in[2];
  const float* w3  = (const float*)d_in[3];

  float* out = (float*)d_out;
  float* o0 = out;                        // 65 x 16384 (pad_tail s0)
  float* o1 = o0 + (size_t)65 * 16384;    // 65 x 4096  (pad_head s1)
  float* o2 = o1 + (size_t)65 * 4096;     // 65 x 4096  (pad_head s2)
  float* o3 = o2 + (size_t)65 * 4096;     // 65 x 1024  (pad_head s3)

  snn_input_spikes<<<16384 / 256, 256, 0, stream>>>(img, o0, 16384);

  // layer1: A = s0 rows 0..63 (same-timestep input), K=16384, N=4096
  snn_layer_wmma<<<4096 / 64, 128, 0, stream>>>(o0, w1, o1, 16384, 4096);
  // layer2: A = [0; s1] rows 0..63 == s1[t-1], K=4096, N=4096
  snn_layer_wmma<<<4096 / 64, 128, 0, stream>>>(o1, w2, o2, 4096, 4096);
  // layer3: A = [0; s2] rows 0..63 == s2[t-1], K=4096, N=1024
  snn_layer_wmma<<<1024 / 64, 128, 0, stream>>>(o2, w3, o3, 4096, 1024);
}
